// TernaryDense_20298015441388
// MI455X (gfx1250) — matrix operand order repaired and checked by reading, not compiled, not run on hardware
//
#include <hip/hip_runtime.h>
#include <stdint.h>

// TernaryDense: out = x @ sign(w) + b
// x: [262144, 128] f32, w: [128, 128] f32, b: [128] f32, out: [262144, 128] f32
//
// Roofline: 8.6 GFLOP vs 268 MB HBM traffic -> memory floor ~11.5 us @ 23.3 TB/s.
// Data movement: x stripe streamed global->LDS with GLOBAL_LOAD_ASYNC_TO_LDS_B128
// (ASYNCcnt path, overlapped with weight ternarization). Matrix path:
// V_WMMA_F32_16X16X32_BF16 (co-executes with VALU as TRANS). Ternary weights are
// exact in bf16; x is split x = hi + lo (two RNE bf16 terms) in-register so each
// product is exact and f32 accumulation recovers ~fp32 accuracy.

typedef float  v4f   __attribute__((ext_vector_type(4)));
typedef float  v8f   __attribute__((ext_vector_type(8)));
typedef __bf16 v16bf __attribute__((ext_vector_type(16)));

#define NROWS    262144
#define KDIM     128
#define NCOLS    128
#define STRIPE   128    // rows per block
#define NTHREADS 256    // 8 waves of 32

__global__ __launch_bounds__(NTHREADS, 3)
void ternary_dense_wmma_bf16_async(const float* __restrict__ x,
                                   const float* __restrict__ w,
                                   const float* __restrict__ bias,
                                   float* __restrict__ out)
{
    extern __shared__ float smem[];
    float*  xs  = smem;                          // [STRIPE][KDIM] f32 x stripe   64 KB
    __bf16* wsb = (__bf16*)(smem + STRIPE*KDIM); // [NCOLS][KDIM]  sign(w)^T bf16 32 KB

    const int tid  = threadIdx.x;
    const int wave = tid >> 5;                   // 0..7 : 16-row tile within stripe
    const int lane = tid & 31;
    const int l15  = lane & 15;
    const int lhi  = lane >> 4;                  // 0 or 1

    const long rowBase = (long)blockIdx.x * STRIPE;

    // ---- kick off async DMA of the x stripe: global -> LDS, 16 B per lane per op.
    // 128*128 f32 = 64 KB = 4096 b128 chunks; 256 threads x 16 iterations.
    {
        const float* xgBase = x + rowBase * KDIM;
        #pragma unroll
        for (int i = 0; i < (STRIPE * KDIM / 4) / NTHREADS; ++i) {
            const int e = (i * NTHREADS + tid) * 4;            // element offset (16 B units)
            const float*   gsrc   = xgBase + e;
            const unsigned dstLds = (unsigned)(uintptr_t)(const void*)(xs + e);
            asm volatile("global_load_async_to_lds_b128 %0, %1, off"
                         :: "v"(dstLds), "v"(gsrc)
                         : "memory");
        }
    }

    // ---- overlap: ternarize weights into LDS, transposed:
    // wsb[n*KDIM + k] = sign(w[k*NCOLS + n]); sign in {-1,0,+1} is exact in bf16.
    for (int i = tid; i < KDIM * NCOLS; i += NTHREADS) {
        const int   k = i >> 7;                  // i / 128
        const int   n = i & 127;                 // i % 128
        const float v = w[i];
        wsb[n * KDIM + k] = (__bf16)((v > 0.0f) ? 1.0f : ((v < 0.0f) ? -1.0f : 0.0f));
    }

    // ---- drain async copies, then publish LDS to all waves.
    asm volatile("s_wait_asynccnt 0" ::: "memory");
    __syncthreads();

    // ---- accumulate: wave owns rows [wave*16, wave*16+16), all 8 column tiles
    v8f acc[8] = {};

    // A (16x32) k order used here (contiguous, same as B): lane l&15 = row M; lanes 0-15 hold K = kc*32 + 0..15,
    // lanes 16-31 hold K = kc*32 + 16..31 -> two contiguous 8-f32 (32 B) chunks.
    const float*  aBase = xs  + (wave * 16 + l15) * KDIM + lhi * 16;
    // B (32x16) k order used here (contiguous, same as A): lane l&15 = col N; lanes 0-15 hold K = kc*32 + 0..15,
    // lanes 16-31 hold K = kc*32 + 16..31 -> one contiguous 16-bf16 (32 B) run in wsb[n][k].
    const __bf16* bBase = wsb + l15 * KDIM + lhi * 16;

    #pragma unroll
    for (int kc = 0; kc < KDIM / 32; ++kc) {     // 4 K-chunks of 32
        const v8f c0 = *(const v8f*)(aBase + kc * 32);        // K = kc*32 + lhi*16 + 0..7
        const v8f c1 = *(const v8f*)(aBase + kc * 32 + 8);    // K = kc*32 + lhi*16 + 8..15
        v16bf ahi, alo;
        #pragma unroll
        for (int j = 0; j < 8; ++j) {
            const __bf16 h0 = (__bf16)c0[j];
            const __bf16 h1 = (__bf16)c1[j];
            ahi[j]     = h0;
            ahi[8 + j] = h1;
            alo[j]     = (__bf16)(c0[j] - (float)h0);
            alo[8 + j] = (__bf16)(c1[j] - (float)h1);
        }
        #pragma unroll
        for (int ct = 0; ct < 8; ++ct) {
            const v16bf bb = *(const v16bf*)(bBase + ct * 16 * KDIM + kc * 32);
            // 8 args: (neg_a, A, neg_b, B, c_mod, C, reuse_a, reuse_b)
            acc[ct] = __builtin_amdgcn_wmma_f32_16x16x32_bf16(
                false, ahi, false, bb, (short)0, acc[ct], false, false);
            acc[ct] = __builtin_amdgcn_wmma_f32_16x16x32_bf16(
                false, alo, false, bb, (short)0, acc[ct], false, false);
        }
    }

    // ---- add bias and store.
    // C layout (32-bit C/D 16x16): VGPR r, lane l -> M = r + 8*(l>=16), N = l&15.
    // Each store: lanes 0-15 contiguous 64 B (row M), lanes 16-31 contiguous 64 B (row M+8).
    #pragma unroll
    for (int ct = 0; ct < 8; ++ct) {
        const float bv = bias[ct * 16 + l15];
        #pragma unroll
        for (int r = 0; r < 8; ++r) {
            const int  m   = r + lhi * 8;
            const long row = rowBase + wave * 16 + m;
            out[row * NCOLS + ct * 16 + l15] = acc[ct][r] + bv;
        }
    }
}

extern "C" void kernel_launch(void* const* d_in, const int* in_sizes, int n_in,
                              void* d_out, int out_size, void* d_ws, size_t ws_size,
                              hipStream_t stream) {
    const float* x = (const float*)d_in[0];
    const float* w = (const float*)d_in[1];
    const float* b = (const float*)d_in[2];
    float* out = (float*)d_out;

    const int nBlocks = NROWS / STRIPE;  // 2048
    const size_t shmem = (size_t)(STRIPE * KDIM) * sizeof(float)
                       + (size_t)(NCOLS * KDIM) * sizeof(__bf16);   // 96 KB
    ternary_dense_wmma_bf16_async<<<nBlocks, NTHREADS, shmem, stream>>>(x, w, b, out);
}
